// BEVMultiHistoryCrossAttention_37374805410492
// MI455X (gfx1250) — matrix operand order repaired and checked by reading, not compiled, not run on hardware
//
#include <hip/hip_runtime.h>
#include <hip/hip_bf16.h>

typedef __attribute__((ext_vector_type(16))) _Float16 v16h;
typedef __attribute__((ext_vector_type(8)))  _Float16 h8;
typedef __attribute__((ext_vector_type(8)))  float    v8f;

#define KDIM 256
#define LDSP (KDIM + 8)   // padded LDS pitch (halves) -> kills bank conflicts

// -----------------------------------------------------------------------------
// C[M,N] = A[M,256] @ B[N,256]^T + bias[N] (+ residual[M,N])
// A,B,bias,residual f32; C stored f32 (Cf) or f16 (Ch) per storeHalf.
// Block: 256 threads = 8 waves; tile 64(M) x 32(N); wave -> one 16x16 WMMA tile
// chain over K. Requires M%64==0, N%32==0 (true for all calls here).
// -----------------------------------------------------------------------------
__global__ __launch_bounds__(256)
void wmma_gemm_atbT(const float* __restrict__ A, const float* __restrict__ B,
                    const float* __restrict__ bias, const float* __restrict__ residual,
                    float* __restrict__ Cf, _Float16* __restrict__ Ch,
                    int M, int N, int storeHalf)
{
    __shared__ _Float16 Ash[64 * LDSP];
    __shared__ _Float16 Bsh[32 * LDSP];

    const int tid  = threadIdx.x;
    const int lane = tid & 31;
    const int wave = tid >> 5;
    const int m0   = blockIdx.x * 64;
    const int n0   = blockIdx.y * 32;

    // Stage A panel 64x256 (f32 -> f16): 4096 float4, 16 per thread
    {
        const float4* Ag = (const float4*)(A + (size_t)m0 * KDIM);
        #pragma unroll
        for (int i = 0; i < 16; ++i) {
            int e    = tid + i * 256;
            int row  = e >> 6;           // 64 float4 per row
            int col4 = e & 63;
            float4 v = Ag[(size_t)row * 64 + col4];
            _Float16* dst = &Ash[row * LDSP + col4 * 4];
            dst[0] = (_Float16)v.x; dst[1] = (_Float16)v.y;
            dst[2] = (_Float16)v.z; dst[3] = (_Float16)v.w;
        }
        // Stage B panel 32x256: 2048 float4, 8 per thread
        const float4* Bg = (const float4*)(B + (size_t)n0 * KDIM);
        #pragma unroll
        for (int i = 0; i < 8; ++i) {
            int e    = tid + i * 256;
            int row  = e >> 6;
            int col4 = e & 63;
            float4 v = Bg[(size_t)row * 64 + col4];
            _Float16* dst = &Bsh[row * LDSP + col4 * 4];
            dst[0] = (_Float16)v.x; dst[1] = (_Float16)v.y;
            dst[2] = (_Float16)v.z; dst[3] = (_Float16)v.w;
        }
    }
    __syncthreads();

    const int msub = wave & 3;            // 4 M-subtiles
    const int nsub = wave >> 2;           // 2 N-subtiles
    const int arow = msub * 16 + (lane & 15);
    const int bcol = nsub * 16 + (lane & 15);
    // A 16x32: lanes 0-15 hold K 0..15, lanes 16-31 hold K 16..31;
    // B 32x16: lanes 0-15 K 0..15, lanes 16-31 K 16..31.
    const int kbaseA = (lane < 16) ? 0 : 16;
    const int kbaseB = (lane < 16) ? 0 : 16;

    v8f acc = {};
    #pragma unroll
    for (int k0 = 0; k0 < KDIM; k0 += 32) {
        union { v16h v; h8 h[2]; } af;
        v16h bfv;
        af.h[0] = *(const h8*) &Ash[arow * LDSP + k0 + kbaseA];
        af.h[1] = *(const h8*) &Ash[arow * LDSP + k0 + kbaseA + 8];
        bfv     = *(const v16h*)&Bsh[bcol * LDSP + k0 + kbaseB];
        acc = __builtin_amdgcn_wmma_f32_16x16x32_f16(
                  /*neg_a=*/false, af.v, /*neg_b=*/false, bfv,
                  /*c_mod=*/(short)0, acc, /*reuse_a=*/false, /*reuse_b=*/false);
    }

    // C/D layout: VGPR i <-> row i (lanes 0-15) or i+8 (lanes 16-31), col = lane%16
    const int rbase = m0 + msub * 16 + ((lane < 16) ? 0 : 8);
    const int gcol  = n0 + bcol;
    const float bv  = bias ? bias[gcol] : 0.0f;
    #pragma unroll
    for (int i = 0; i < 8; ++i) {
        int    grow = rbase + i;
        size_t cidx = (size_t)grow * N + gcol;
        float  val  = acc[i] + bv;
        if (residual) val += residual[cidx];
        if (storeHalf) Ch[cidx] = (_Float16)val;
        else           Cf[cidx] = val;
    }
}

// -----------------------------------------------------------------------------
// Deformable sampling: wave per (q, head), lane per channel (HD=32).
// Fuses softmax over the 8 points, bilinear gather from f16 v-proj (L2-resident),
// weighted sum, and the mean over the R=2 history batch.
// vproj: f16 [2][16384][8][32]; off: f32 [16384][8][8][2]; awl: f32 [16384][8][8];
// refp: f32 [2][16384][4][2]; outacc: f32 [16384][256]
// -----------------------------------------------------------------------------
__device__ __forceinline__ float sample_corner(const _Float16* __restrict__ vb,
                                               int iy, int ix)
{
    bool valid = (iy >= 0) & (iy < 128) & (ix >= 0) & (ix < 128);
    int  cy = min(max(iy, 0), 127);
    int  cx = min(max(ix, 0), 127);
    float g = (float)vb[(size_t)(cy * 128 + cx) * 256];  // spatial stride = NH*HD
    return valid ? g : 0.0f;
}

__global__ __launch_bounds__(256)
void deform_sample(const _Float16* __restrict__ vproj,
                   const float* __restrict__ off,
                   const float* __restrict__ awl,
                   const float* __restrict__ refp,
                   float* __restrict__ outacc)
{
    const int tid  = blockIdx.x * blockDim.x + threadIdx.x;
    const int wid  = tid >> 5;
    const int lane = tid & 31;
    const int q    = wid >> 3;
    const int h    = wid & 7;
    if (q >= 16384) return;

    // softmax over 8 logits (uniform across lanes; cheap)
    float lg[8];
    const float* ap = awl + ((size_t)q * 8 + h) * 8;
    float mx = -1e30f;
    #pragma unroll
    for (int p = 0; p < 8; ++p) { lg[p] = ap[p]; mx = fmaxf(mx, lg[p]); }
    float s = 0.0f;
    #pragma unroll
    for (int p = 0; p < 8; ++p) { lg[p] = __expf(lg[p] - mx); s += lg[p]; }
    const float inv = 1.0f / s;

    const float* op = off + ((size_t)q * 8 + h) * 16;   // [p][2]
    float acc = 0.0f;

    #pragma unroll
    for (int r = 0; r < 2; ++r) {
        const float*    rp = refp  + ((size_t)r * 16384 + q) * 8;        // [za=4][2]
        const _Float16* vb = vproj + (size_t)r * 16384 * 256 + h * 32 + lane;
        #pragma unroll
        for (int p = 0; p < 8; ++p) {
            int   z  = p & 3;                         // p = i*ZA + z, ZA=4
            float lx = rp[z * 2 + 0] + op[p * 2 + 0] * (1.0f / 128.0f);
            float ly = rp[z * 2 + 1] + op[p * 2 + 1] * (1.0f / 128.0f);
            float x  = lx * 128.0f - 0.5f;
            float y  = ly * 128.0f - 0.5f;
            float x0f = floorf(x), y0f = floorf(y);
            float wx = x - x0f, wy = y - y0f;
            int   x0 = (int)x0f, y0 = (int)y0f;
            float g00 = sample_corner(vb, y0,     x0);
            float g01 = sample_corner(vb, y0,     x0 + 1);
            float g10 = sample_corner(vb, y0 + 1, x0);
            float g11 = sample_corner(vb, y0 + 1, x0 + 1);
            float bil = (g00 * (1.0f - wx) + g01 * wx) * (1.0f - wy)
                      + (g10 * (1.0f - wx) + g11 * wx) * wy;
            acc += lg[p] * inv * bil;
        }
    }
    outacc[(size_t)q * 256 + h * 32 + lane] = acc * 0.5f;  // mean over R=2
}

// -----------------------------------------------------------------------------
extern "C" void kernel_launch(void* const* d_in, const int* in_sizes, int n_in,
                              void* d_out, int out_size, void* d_ws, size_t ws_size,
                              hipStream_t stream)
{
    const float* query  = (const float*)d_in[0];   // [1,16384,256]
    const float* value  = (const float*)d_in[1];   // [2,16384,256]
    const float* refp   = (const float*)d_in[2];   // [2,16384,4,2]
    // d_in[3] spatial_shapes == [[128,128]] (hardcoded)
    const float* W_off  = (const float*)d_in[4];   // [128,256]
    const float* b_off  = (const float*)d_in[5];
    const float* W_attn = (const float*)d_in[6];   // [64,256]
    const float* b_attn = (const float*)d_in[7];
    const float* W_val  = (const float*)d_in[8];   // [256,256]
    const float* b_val  = (const float*)d_in[9];
    const float* W_out  = (const float*)d_in[10];  // [256,256]
    const float* b_out  = (const float*)d_in[11];
    float* out = (float*)d_out;                    // [1,16384,256]

    char* ws = (char*)d_ws;
    _Float16* vproj = (_Float16*)(ws);                     // 16 MB  f16 [2][16384][8][32]
    float*    offb  = (float*)(ws + (16ull << 20));        //  8 MB  [16384][128]
    float*    awlb  = (float*)(ws + (24ull << 20));        //  4 MB  [16384][64]
    float*    outac = (float*)(ws + (28ull << 20));        // 16 MB  [16384][256]

    // 1) v-proj (f16 output -> sampling footprint fits L2)
    wmma_gemm_atbT<<<dim3(32768 / 64, 256 / 32), 256, 0, stream>>>(
        value, W_val, b_val, nullptr, nullptr, vproj, 32768, 256, 1);
    // 2) sampling offsets (computed ONCE: query identical across R histories)
    wmma_gemm_atbT<<<dim3(16384 / 64, 128 / 32), 256, 0, stream>>>(
        query, W_off, b_off, nullptr, offb, nullptr, 16384, 128, 0);
    // 3) attention logits (softmax fused into sampling kernel)
    wmma_gemm_atbT<<<dim3(16384 / 64, 64 / 32), 256, 0, stream>>>(
        query, W_attn, b_attn, nullptr, awlb, nullptr, 16384, 64, 0);
    // 4) softmax + bilinear gather + weighted sum + history mean
    deform_sample<<<dim3(16384), 256, 0, stream>>>(vproj, offb, awlb, refp, outac);
    // 5) output proj + bias + residual, straight into d_out
    wmma_gemm_atbT<<<dim3(16384 / 64, 256 / 32), 256, 0, stream>>>(
        outac, W_out, b_out, query, out, nullptr, 16384, 256, 0);
}
